// MoETransformer_1769526526371
// MI455X (gfx1250) — matrix operand order repaired and checked by reading, not compiled, not run on hardware
//
#include <hip/hip_runtime.h>
#include <hip/hip_bf16.h>

typedef __attribute__((ext_vector_type(16))) __bf16 v16bf;
typedef __attribute__((ext_vector_type(8)))  float  v8f;

using u32 = unsigned int;
using u16 = unsigned short;

#define N_TOK 8192
#define DIM_D 1024
#define DIM_H 2048
#define DIM_O 1024
#define NEXP  8
#define DIM_G 256
#define MAXR  16896   // N_TOK*2 assignments + NEXP*64 padding, multiple of 64

__device__ __forceinline__ u16 f2bf(float f) {
  u32 u = __float_as_uint(f);
  u32 r = u + 0x7FFFu + ((u >> 16) & 1u);   // round-to-nearest-even
  return (u16)(r >> 16);
}

union Frag { u32 u[8]; v16bf v; };

// ---------------------------------------------------------------- init
__global__ __launch_bounds__(256) void init_kernel(int* counts, int* atok) {
  int i = blockIdx.x * 256 + threadIdx.x;
  if (i < NEXP) counts[i] = 0;
  if (i < MAXR) atok[i] = -1;
}

// ---------------------------------------------------------------- gating
// One wave per 16 tokens: hgate = relu(x@Wg1+bg1) via WMMA, then logits,
// softmax, top-2, renormalized weights, usage counts.
__global__ __launch_bounds__(32) void gate_kernel(
    const float* __restrict__ x, const float* __restrict__ Wg1,
    const float* __restrict__ bg1, const float* __restrict__ Wg2,
    const float* __restrict__ bg2, int* __restrict__ sel,
    float* __restrict__ wsel, int* __restrict__ counts) {
  __shared__ __align__(16) u16 lX[16 * DIM_D];
  __shared__ __align__(16) u16 lB[16 * 36];
  __shared__ float hg[16 * DIM_G];

  const int lane = threadIdx.x;
  const int t0 = blockIdx.x * 16;

  // stage 16x1024 x-tile as bf16 in LDS
  for (int i = lane; i < 16 * DIM_D / 4; i += 32) {
    int r = i >> 8;
    int c = (i & 255) * 4;
    float4 xv = *(const float4*)(x + (size_t)(t0 + r) * DIM_D + c);
    lX[r * DIM_D + c + 0] = f2bf(xv.x);
    lX[r * DIM_D + c + 1] = f2bf(xv.y);
    lX[r * DIM_D + c + 2] = f2bf(xv.z);
    lX[r * DIM_D + c + 3] = f2bf(xv.w);
  }
  __syncthreads();

  const int cs = lane & 15;
  const int hi = (lane < 16) ? 0 : 1;

  for (int nt = 0; nt < DIM_G / 16; nt++) {
    const int nBase = nt * 16;
    v8f acc = {0.f, 0.f, 0.f, 0.f, 0.f, 0.f, 0.f, 0.f};
    for (int ks = 0; ks < DIM_D / 32; ks++) {
      __syncthreads();
      // Wg1 tile 32x16 -> n-major LDS (padded stride 36)
      for (int i = 0; i < 16; i++) {
        int k = hi * 16 + i;
        lB[cs * 36 + k] = f2bf(Wg1[(size_t)(ks * 32 + k) * DIM_G + nBase + cs]);
      }
      __syncthreads();
      Frag a, b;
      const u32* pa = (const u32*)lX + cs * (DIM_D / 2) + ks * 16 + hi * 4;
#pragma unroll
      for (int i = 0; i < 4; i++) { a.u[i] = pa[i]; a.u[4 + i] = pa[8 + i]; }
      const u32* pb = (const u32*)lB + cs * 18 + hi * 4;
#pragma unroll
      for (int i = 0; i < 4; i++) { b.u[i] = pb[i]; b.u[4 + i] = pb[8 + i]; }
      acc = __builtin_amdgcn_wmma_f32_16x16x32_bf16(
          false, a.v, false, b.v, (short)0, acc, false, false);
    }
    float bias = bg1[nBase + cs];
    int rb = hi * 8;
#pragma unroll
    for (int r = 0; r < 8; r++)
      hg[(rb + r) * DIM_G + nBase + cs] = fmaxf(acc[r] + bias, 0.f);
  }
  __syncthreads();

  if (lane < 16) {
    const int tok = t0 + lane;
    float l[NEXP];
    for (int e = 0; e < NEXP; e++) {
      float s = bg2[e];
      for (int g = 0; g < DIM_G; g++) s += hg[lane * DIM_G + g] * Wg2[g * NEXP + e];
      l[e] = s;
    }
    int a0 = 0;
    for (int e = 1; e < NEXP; e++) if (l[e] > l[a0]) a0 = e;   // ties -> lowest idx
    int a1 = (a0 == 0) ? 1 : 0;
    for (int e = 0; e < NEXP; e++)
      if (e != a0 && l[e] > l[a1]) a1 = e;
    float e1 = __expf(l[a1] - l[a0]);
    float inv = 1.f / (1.f + e1);                // w0 = p0/(p0+p1), w1 = p1/(p0+p1)
    sel[tok * 2] = a0; sel[tok * 2 + 1] = a1;
    wsel[tok * 2] = inv; wsel[tok * 2 + 1] = e1 * inv;
    atomicAdd(&counts[a0], 1);
    atomicAdd(&counts[a1], 1);
  }
}

// ---------------------------------------------------------------- scan
__global__ void scan_kernel(const int* counts, int* offs, int* cursors, int* rtot) {
  if (threadIdx.x == 0) {
    int o = 0;
    for (int e = 0; e < NEXP; e++) {
      offs[e] = o; cursors[e] = o;
      o += ((counts[e] + 63) >> 6) << 6;         // pad each expert to 64 rows
    }
    offs[NEXP] = o;
    *rtot = o;
  }
}

// ---------------------------------------------------------------- scatter
__global__ __launch_bounds__(256) void scatter_kernel(
    const int* __restrict__ sel, const float* __restrict__ wsel,
    int* __restrict__ cursors, int* __restrict__ atok,
    float* __restrict__ aw, int* __restrict__ slot) {
  int t = blockIdx.x * 256 + threadIdx.x;
  if (t >= N_TOK) return;
  for (int k = 0; k < 2; k++) {
    int e = sel[2 * t + k];
    int i = atomicAdd(&cursors[e], 1);
    atok[i] = t;
    aw[i] = wsel[2 * t + k];
    slot[2 * t + k] = i;
  }
}

// ---------------------------------------------------------------- gather x -> bf16
__global__ __launch_bounds__(256) void gather_kernel(
    const float* __restrict__ x, const int* __restrict__ atok,
    u16* __restrict__ Abf) {
  const int row = blockIdx.x;
  const int c = threadIdx.x * 4;
  int tok = atok[row];
  u32 p0 = 0, p1 = 0;
  if (tok >= 0) {
    float4 xv = *(const float4*)(x + (size_t)tok * DIM_D + c);
    p0 = (u32)f2bf(xv.x) | ((u32)f2bf(xv.y) << 16);
    p1 = (u32)f2bf(xv.z) | ((u32)f2bf(xv.w) << 16);
  }
  u32* dst = (u32*)(Abf + (size_t)row * DIM_D + c);
  dst[0] = p0; dst[1] = p1;
}

// ---------------------------------------------------------------- expert GEMM
// BM=64 x BN=256 x BK=32, 8 waves (2x4), each wave a 32x64 C block
// (2 A-frags x 4 B-frags = 8 WMMAs per K-step). Depth-1 software pipeline:
// K-step ks+1 global->regs overlaps WMMA of step ks. A: bf16 [rows,Kd];
// W: f32 [E][Kd][Nd] converted on the fly to bf16 in n-major LDS.
__global__ __launch_bounds__(256) void gemm_bf16(
    const u16* __restrict__ A, const float* __restrict__ Wall,
    const float* __restrict__ ball, int Kd, int Nd,
    const int* __restrict__ offs, const int* __restrict__ rtot,
    const float* __restrict__ aw, u16* __restrict__ outBf,
    float* __restrict__ outF, int doRelu) {
  __shared__ __align__(16) u16 lA[64 * 32];
  __shared__ __align__(16) u16 lB[256 * 36];

  const int rowBase = blockIdx.y * 64;
  if (rowBase >= *rtot) return;
  const int nBase = blockIdx.x * 256;

  int e = 0;
  for (int i = 1; i <= NEXP; i++)
    if (rowBase >= offs[i]) e = i;               // tiles never straddle experts

  const float* W = Wall + (size_t)e * Kd * Nd;
  const float* b = ball + (size_t)e * Nd;

  const int lane = threadIdx.x & 31;
  const int wid = threadIdx.x >> 5;
  const int rowOff = (wid & 1) * 32;
  const int nOff = (wid >> 1) * 64;
  const int cs = lane & 15;
  const int hi = (lane < 16) ? 0 : 1;

  v8f c[2][4];
#pragma unroll
  for (int m = 0; m < 2; m++)
#pragma unroll
    for (int j = 0; j < 4; j++)
#pragma unroll
      for (int r = 0; r < 8; r++) c[m][j][r] = 0.f;

  const int aR = threadIdx.x >> 2;         // 0..63
  const int aC = (threadIdx.x & 3) * 8;    // 0..24
  const int bN = (threadIdx.x & 63) * 4;   // 0..252
  const int bK0 = threadIdx.x >> 6;        // 0..3

  const int ksteps = Kd / 32;
  uint4 aReg;
  float4 wReg[8];

  // prologue: fetch tile 0
  aReg = *(const uint4*)(A + (size_t)(rowBase + aR) * Kd + aC);
#pragma unroll
  for (int kk = 0; kk < 8; kk++)
    wReg[kk] = *(const float4*)(W + (size_t)(bK0 + kk * 4) * Nd + nBase + bN);
  *(uint4*)(lA + aR * 32 + aC) = aReg;
#pragma unroll
  for (int kk = 0; kk < 8; kk++) {
    int k = bK0 + kk * 4;
    lB[(bN + 0) * 36 + k] = f2bf(wReg[kk].x);
    lB[(bN + 1) * 36 + k] = f2bf(wReg[kk].y);
    lB[(bN + 2) * 36 + k] = f2bf(wReg[kk].z);
    lB[(bN + 3) * 36 + k] = f2bf(wReg[kk].w);
  }

  for (int ks = 0; ks < ksteps; ks++) {
    __syncthreads();                        // LDS tile for `ks` visible

    if (ks + 1 < ksteps) {                  // overlap next-tile fetch with WMMA
      aReg = *(const uint4*)(A + (size_t)(rowBase + aR) * Kd + (ks + 1) * 32 + aC);
#pragma unroll
      for (int kk = 0; kk < 8; kk++)
        wReg[kk] = *(const float4*)(W + (size_t)((ks + 1) * 32 + bK0 + kk * 4) * Nd +
                                    nBase + bN);
    }
    if (ks + 2 < ksteps)                    // gfx1250 prefetch of HBM A stream
      __builtin_prefetch(A + (size_t)(rowBase + aR) * Kd + (ks + 2) * 32 + aC, 0, 1);

    Frag a[2], bfr[4];
#pragma unroll
    for (int m = 0; m < 2; m++) {
      const u32* pa = (const u32*)lA + (rowOff + m * 16 + cs) * 16 + hi * 4;
#pragma unroll
      for (int i = 0; i < 4; i++) { a[m].u[i] = pa[i]; a[m].u[4 + i] = pa[8 + i]; }
    }
#pragma unroll
    for (int j = 0; j < 4; j++) {
      const u32* pb = (const u32*)lB + (nOff + j * 16 + cs) * 18 + hi * 4;
#pragma unroll
      for (int i = 0; i < 4; i++) { bfr[j].u[i] = pb[i]; bfr[j].u[4 + i] = pb[8 + i]; }
    }
#pragma unroll
    for (int m = 0; m < 2; m++)
#pragma unroll
      for (int j = 0; j < 4; j++)
        c[m][j] = __builtin_amdgcn_wmma_f32_16x16x32_bf16(
            false, a[m].v, false, bfr[j].v, (short)0, c[m][j], false, false);

    __syncthreads();                        // everyone done reading LDS

    if (ks + 1 < ksteps) {                  // stage next tile into LDS
      *(uint4*)(lA + aR * 32 + aC) = aReg;
#pragma unroll
      for (int kk = 0; kk < 8; kk++) {
        int k = bK0 + kk * 4;
        lB[(bN + 0) * 36 + k] = f2bf(wReg[kk].x);
        lB[(bN + 1) * 36 + k] = f2bf(wReg[kk].y);
        lB[(bN + 2) * 36 + k] = f2bf(wReg[kk].z);
        lB[(bN + 3) * 36 + k] = f2bf(wReg[kk].w);
      }
    }
  }

  // epilogue: +bias, optional relu -> bf16, or gate-weight scale -> f32
  const int rb = hi * 8;
#pragma unroll
  for (int m = 0; m < 2; m++) {
#pragma unroll
    for (int j = 0; j < 4; j++) {
      int nG = nBase + nOff + j * 16 + cs;
      float bias = b[nG];
#pragma unroll
      for (int r = 0; r < 8; r++) {
        int rG = rowBase + rowOff + m * 16 + rb + r;
        float v = c[m][j][r] + bias;
        if (doRelu) v = fmaxf(v, 0.f);
        if (outBf) outBf[(size_t)rG * Nd + nG] = f2bf(v);
        else       outF[(size_t)rG * Nd + nG] = v * aw[rG];
      }
    }
  }
}

// ---------------------------------------------------------------- combine
__global__ __launch_bounds__(256) void combine_kernel(
    const float* __restrict__ tmp, const int* __restrict__ slot,
    float* __restrict__ out) {
  const int t = blockIdx.x;
  const int c = threadIdx.x * 4;
  int s0 = slot[2 * t], s1 = slot[2 * t + 1];
  float4 a = *(const float4*)(tmp + (size_t)s0 * DIM_O + c);
  float4 b = *(const float4*)(tmp + (size_t)s1 * DIM_O + c);
  float4 o;
  o.x = a.x + b.x; o.y = a.y + b.y; o.z = a.z + b.z; o.w = a.w + b.w;
  *(float4*)(out + (size_t)t * DIM_O + c) = o;
}

// ---------------------------------------------------------------- aux losses
__global__ void loss_kernel(const int* counts, float* out) {
  if (threadIdx.x == 0) {
    float s = 0.f;
    for (int e = 0; e < NEXP; e++) {
      float u = (float)counts[e] / (float)N_TOK;
      out[1 + e] = u;
      float d = u - 1.f / NEXP;
      s += d * d;
    }
    out[0] = (s / NEXP) * 0.01f;
  }
}

// ---------------------------------------------------------------- launch
extern "C" void kernel_launch(void* const* d_in, const int* in_sizes, int n_in,
                              void* d_out, int out_size, void* d_ws, size_t ws_size,
                              hipStream_t stream) {
  (void)in_sizes; (void)n_in; (void)out_size; (void)ws_size;
  const float* x   = (const float*)d_in[0];
  const float* Wg1 = (const float*)d_in[1];
  const float* bg1 = (const float*)d_in[2];
  const float* Wg2 = (const float*)d_in[3];
  const float* bg2 = (const float*)d_in[4];
  const float* W1  = (const float*)d_in[5];
  const float* b1  = (const float*)d_in[6];
  const float* W2  = (const float*)d_in[7];
  const float* b2  = (const float*)d_in[8];
  const float* W3  = (const float*)d_in[9];
  const float* b3  = (const float*)d_in[10];
  float* out = (float*)d_out;

  char* w = (char*)d_ws;
  size_t off = 0;
  auto alloc = [&](size_t bytes) {
    size_t o = off;
    off += (bytes + 255) & ~(size_t)255;
    return o;
  };
  int*   counts  = (int*)(w + alloc(NEXP * 4));
  int*   cursors = (int*)(w + alloc(NEXP * 4));
  int*   offs    = (int*)(w + alloc((NEXP + 1) * 4));
  int*   rtot    = (int*)(w + alloc(4));
  int*   sel     = (int*)(w + alloc((size_t)N_TOK * 2 * 4));
  float* wsel    = (float*)(w + alloc((size_t)N_TOK * 2 * 4));
  int*   atok    = (int*)(w + alloc((size_t)MAXR * 4));
  float* aw      = (float*)(w + alloc((size_t)MAXR * 4));
  int*   slot    = (int*)(w + alloc((size_t)N_TOK * 2 * 4));
  u16*   Abf     = (u16*)(w + alloc((size_t)MAXR * DIM_D * 2));
  u16*   h1      = (u16*)(w + alloc((size_t)MAXR * DIM_H * 2));
  u16*   h2      = (u16*)(w + alloc((size_t)MAXR * DIM_H * 2));
  float* tmp     = (float*)(w + alloc((size_t)MAXR * DIM_O * 4));

  init_kernel<<<(MAXR + 255) / 256, 256, 0, stream>>>(counts, atok);
  gate_kernel<<<N_TOK / 16, 32, 0, stream>>>(x, Wg1, bg1, Wg2, bg2, sel, wsel, counts);
  scan_kernel<<<1, 1, 0, stream>>>(counts, offs, cursors, rtot);
  scatter_kernel<<<N_TOK / 256, 256, 0, stream>>>(sel, wsel, cursors, atok, aw, slot);
  gather_kernel<<<MAXR, 256, 0, stream>>>(x, atok, Abf);

  gemm_bf16<<<dim3(DIM_H / 256, MAXR / 64), 256, 0, stream>>>(
      Abf, W1, b1, DIM_D, DIM_H, offs, rtot, aw, h1, nullptr, 1);
  gemm_bf16<<<dim3(DIM_H / 256, MAXR / 64), 256, 0, stream>>>(
      h1, W2, b2, DIM_H, DIM_H, offs, rtot, aw, h2, nullptr, 1);
  gemm_bf16<<<dim3(DIM_O / 256, MAXR / 64), 256, 0, stream>>>(
      h2, W3, b3, DIM_H, DIM_O, offs, rtot, aw, nullptr, tmp, 0);

  combine_kernel<<<N_TOK, 256, 0, stream>>>(tmp, slot, out);
  loss_kernel<<<1, 32, 0, stream>>>(counts, out + (size_t)N_TOK * DIM_O);
}
